// RBF_56298431316600
// MI455X (gfx1250) — matrix operand order repaired and checked by reading, not compiled, not run on hardware
//
#include <hip/hip_runtime.h>
#include <stdint.h>

// ---------------------------------------------------------------------------
// RBF layer: out[n][j] = exp(-beta[j] * sqrt(max(||x_n||^2 + ||c_j||^2 - 2 x_n.c_j, 0)))
// Strategy (MI455X / gfx1250):
//   * prep pass: f32 -> (bf16 hi, bf16 lo) split planes + row sum-of-squares
//     (NT loads: sources are read exactly once, keep L2 for the planes)
//   * GEMM: 3-term bf16 WMMA emulation of f32 (xh*ch + xh*cl + xl*ch),
//     f32 accumulation via v_wmma_f32_16x16x32_bf16
//   * LDS double-buffered tiles filled with global_load_async_to_lds_b128
//     (ASYNCcnt + s_wait_asynccnt), fused sqrt/exp epilogue with NT stores
//     (256 MB output stream must not evict the 80 MB bf16 planes from L2).
// ---------------------------------------------------------------------------

typedef __attribute__((ext_vector_type(16))) __bf16 v16bf;
typedef __attribute__((ext_vector_type(8)))  float  v8f;

union Frag32B { uint4 q[2]; v16bf v; };

__device__ __forceinline__ unsigned bf16_rne(float f) {
  unsigned u = __float_as_uint(f);
  return (u + 0x7FFFu + ((u >> 16) & 1u)) >> 16;   // round-to-nearest-even
}

// ------------------------- prep kernel -------------------------------------
__global__ __launch_bounds__(256)
void rbf_prep(const float* __restrict__ src,
              unsigned short* __restrict__ hi,
              unsigned short* __restrict__ lo,
              float* __restrict__ sq, int cols) {
  const int row = blockIdx.x;
  const float* r = src + (size_t)row * cols;
  unsigned short* hp = hi + (size_t)row * cols;
  unsigned short* lp = lo + (size_t)row * cols;
  float s = 0.f;
  for (int i = threadIdx.x; i < cols; i += 256) {
    float f = __builtin_nontemporal_load(&r[i]);   // single-use stream
    s = fmaf(f, f, s);
    unsigned hb = bf16_rne(f);
    float hf = __uint_as_float(hb << 16);
    unsigned lb = bf16_rne(f - hf);                // residual plane
    hp[i] = (unsigned short)hb;
    lp[i] = (unsigned short)lb;
  }
  __shared__ float red[256];
  red[threadIdx.x] = s;
  __syncthreads();
  for (int off = 128; off > 0; off >>= 1) {
    if ((int)threadIdx.x < off) red[threadIdx.x] += red[threadIdx.x + off];
    __syncthreads();
  }
  if (threadIdx.x == 0) sq[row] = red[0];
}

// ------------------------- main GEMM kernel --------------------------------
#define BM 64
#define BN 128
#define BK 32
#define KD 1024
#define ND 4096
#define KTILES (KD / BK)

#define ROWB 80                       // LDS row pitch bytes: 32 bf16 + 8 pad
#define AH_OFF 0
#define AL_OFF (BM * ROWB)            // 5120
#define BH_OFF (2 * BM * ROWB)        // 10240
#define BL_OFF (2 * BM * ROWB + BN * ROWB)  // 20480
#define BUFB   (2 * BM * ROWB + 2 * BN * ROWB)  // 30720 bytes / buffer

#define ASYNC_B128(ldsoff, goff, base64)                                  \
  asm volatile("global_load_async_to_lds_b128 %0, %1, %2"                 \
               :: "v"(ldsoff), "v"(goff), "s"(base64) : "memory")

__global__ __launch_bounds__(256)
void rbf_wmma(const unsigned short* __restrict__ xhi,
              const unsigned short* __restrict__ xlo,
              const unsigned short* __restrict__ chi,
              const unsigned short* __restrict__ clo,
              const float* __restrict__ xsq,
              const float* __restrict__ csq,
              const float* __restrict__ betas,
              float* __restrict__ out) {
  __shared__ __align__(128) unsigned char smem[2 * BUFB];
  const uint32_t lds_base = (uint32_t)(uintptr_t)smem;

  const int tid   = threadIdx.x;
  const int lane  = tid & 31;
  const int wave  = tid >> 5;      // 8 waves
  const int wm    = wave & 1;      // 2 M sub-blocks of 32
  const int wn    = wave >> 1;     // 4 N sub-blocks of 32
  const int lrow  = lane & 15;
  const int khalf = lane >> 4;

  const int m0 = blockIdx.y * BM;
  const int n0 = blockIdx.x * BN;

  // ---- async-copy indexing: one b128 (8 bf16) chunk per lane-op -----------
  const int arow = tid >> 2;             // A: 64 rows x 4 chunks
  const int akc  = tid & 3;
  const uint32_t a_goff = (uint32_t)((m0 + arow) * KD + akc * 8) * 2u;
  const uint32_t a_lds  = (uint32_t)(arow * ROWB + akc * 16);

  const int brow0 = tid >> 2;            // B: 128 rows x 4 chunks, 2 per thread
  const int brow1 = brow0 + 64;
  const int bkc   = tid & 3;
  const uint32_t b_goff0 = (uint32_t)((n0 + brow0) * KD + bkc * 8) * 2u;
  const uint32_t b_goff1 = (uint32_t)((n0 + brow1) * KD + bkc * 8) * 2u;
  const uint32_t b_lds0  = (uint32_t)(brow0 * ROWB + bkc * 16);
  const uint32_t b_lds1  = (uint32_t)(brow1 * ROWB + bkc * 16);

  const uint64_t xh64 = (uint64_t)(uintptr_t)xhi;
  const uint64_t xl64 = (uint64_t)(uintptr_t)xlo;
  const uint64_t ch64 = (uint64_t)(uintptr_t)chi;
  const uint64_t cl64 = (uint64_t)(uintptr_t)clo;

  auto issue = [&](int kt, int bufq) {
    const uint32_t kb = (uint32_t)kt * (BK * 2);      // K step in bytes
    const uint32_t bb = lds_base + (uint32_t)bufq * BUFB;
    ASYNC_B128(bb + AH_OFF + a_lds,  a_goff  + kb, xh64);
    ASYNC_B128(bb + AL_OFF + a_lds,  a_goff  + kb, xl64);
    ASYNC_B128(bb + BH_OFF + b_lds0, b_goff0 + kb, ch64);
    ASYNC_B128(bb + BH_OFF + b_lds1, b_goff1 + kb, ch64);
    ASYNC_B128(bb + BL_OFF + b_lds0, b_goff0 + kb, cl64);
    ASYNC_B128(bb + BL_OFF + b_lds1, b_goff1 + kb, cl64);
  };

  v8f acc[2][2];
  const v8f vzero = {0.f, 0.f, 0.f, 0.f, 0.f, 0.f, 0.f, 0.f};
#pragma unroll
  for (int mt = 0; mt < 2; ++mt)
#pragma unroll
    for (int nt = 0; nt < 2; ++nt) acc[mt][nt] = vzero;

  issue(0, 0);
  int buf = 0;
  for (int kt = 0; kt < KTILES; ++kt) {
    if (kt + 1 < KTILES) {
      issue(kt + 1, buf ^ 1);
      asm volatile("s_wait_asynccnt 0x6" ::: "memory");  // tile kt complete
    } else {
      asm volatile("s_wait_asynccnt 0x0" ::: "memory");
    }
    __syncthreads();   // tile kt visible from ALL waves' async copies

    const unsigned char* bufp = smem + buf * BUFB;

    Frag32B ah[2], al[2], bh[2], bl[2];
#pragma unroll
    for (int mt = 0; mt < 2; ++mt) {
      const int ml = wm * 32 + mt * 16 + lrow;
      const unsigned char* ph = bufp + AH_OFF + ml * ROWB + khalf * 16;
      ah[mt].q[0] = *(const uint4*)(ph);        // K = khalf*8 .. +7
      ah[mt].q[1] = *(const uint4*)(ph + 32);   // K = 16+khalf*8 .. +7
      const unsigned char* pl = bufp + AL_OFF + ml * ROWB + khalf * 16;
      al[mt].q[0] = *(const uint4*)(pl);
      al[mt].q[1] = *(const uint4*)(pl + 32);
    }
#pragma unroll
    for (int nt = 0; nt < 2; ++nt) {
      const int nl = wn * 32 + nt * 16 + lrow;
      const unsigned char* ph = bufp + BH_OFF + nl * ROWB + khalf * 16;
      bh[nt].q[0] = *(const uint4*)(ph);        // K = khalf*8 .. +7
      bh[nt].q[1] = *(const uint4*)(ph + 32);
      const unsigned char* pl = bufp + BL_OFF + nl * ROWB + khalf * 16;
      bl[nt].q[0] = *(const uint4*)(pl);
      bl[nt].q[1] = *(const uint4*)(pl + 32);
    }

#pragma unroll
    for (int mt = 0; mt < 2; ++mt)
#pragma unroll
      for (int nt = 0; nt < 2; ++nt) {
        acc[mt][nt] = __builtin_amdgcn_wmma_f32_16x16x32_bf16(
            false, ah[mt].v, false, bh[nt].v, (short)0, acc[mt][nt], false, false);
        acc[mt][nt] = __builtin_amdgcn_wmma_f32_16x16x32_bf16(
            false, ah[mt].v, false, bl[nt].v, (short)0, acc[mt][nt], false, false);
        acc[mt][nt] = __builtin_amdgcn_wmma_f32_16x16x32_bf16(
            false, al[mt].v, false, bh[nt].v, (short)0, acc[mt][nt], false, false);
      }

    __syncthreads();   // done reading buf before it is refilled
    buf ^= 1;
  }

  // ---- epilogue: d2 = ||x||^2 + ||c||^2 - 2*dot;  out = exp(-beta*sqrt(d2))
#pragma unroll
  for (int mt = 0; mt < 2; ++mt) {
    const int rbase = m0 + wm * 32 + mt * 16 + khalf * 8;
    float xs[8];
#pragma unroll
    for (int i = 0; i < 8; ++i) xs[i] = xsq[rbase + i];
#pragma unroll
    for (int nt = 0; nt < 2; ++nt) {
      const int col = n0 + wn * 32 + nt * 16 + lrow;
      const float cs = csq[col];
      const float be = betas[col];
#pragma unroll
      for (int i = 0; i < 8; ++i) {
        const int row = rbase + i;
        float d2 = fmaxf(xs[i] + cs - 2.f * acc[mt][nt][i], 0.f);
        float v  = __expf(-be * sqrtf(d2));
        __builtin_nontemporal_store(v, &out[(size_t)row * ND + col]);
      }
    }
  }
}

// ------------------------- launcher ----------------------------------------
extern "C" void kernel_launch(void* const* d_in, const int* in_sizes, int n_in,
                              void* d_out, int out_size, void* d_ws, size_t ws_size,
                              hipStream_t stream) {
  const float* x       = (const float*)d_in[0];   // [16384,1024]
  const float* centers = (const float*)d_in[1];   // [4096,1024]
  const float* betas   = (const float*)d_in[2];   // [4096]
  float* out = (float*)d_out;                     // [16384,4096]

  const int NROWS = in_sizes[0] / KD;             // 16384
  const int NOUT  = in_sizes[2];                  // 4096

  uint8_t* ws = (uint8_t*)d_ws;
  float* xsq = (float*)ws;                                   // 16384 f32
  float* csq = (float*)(ws + 65536);                         // 4096 f32
  unsigned short* xhi = (unsigned short*)(ws + 81920);       // 16384x1024 bf16
  unsigned short* xlo = xhi + (size_t)NROWS * KD;
  unsigned short* chi = xlo + (size_t)NROWS * KD;
  unsigned short* clo = chi + (size_t)NOUT * KD;

  rbf_prep<<<NROWS, 256, 0, stream>>>(x, xhi, xlo, xsq, KD);
  rbf_prep<<<NOUT,  256, 0, stream>>>(centers, chi, clo, csq, KD);

  dim3 grid(NOUT / BN, NROWS / BM);   // (32, 256)
  rbf_wmma<<<grid, 256, 0, stream>>>(xhi, xlo, chi, clo, xsq, csq, betas, out);
}
